// NonLocalBlock_2_20633022890685
// MI455X (gfx1250) — compile-verified, matrix operand order checked by reading, not run on hardware
//
#include <hip/hip_runtime.h>
#include <hip/hip_bf16.h>
#include <cstdint>

typedef __attribute__((ext_vector_type(16))) _Float16 v16h;
typedef __attribute__((ext_vector_type(8)))  float    v8f;

#define BB   4
#define CINX 256
#define FFX  128
#define NNX  4096   // H*W = 64*64

// ---- CDNA5 async global->LDS copy (ASYNCcnt-tracked), with safe fallback ----
#if defined(__gfx1250__) && \
    __has_builtin(__builtin_amdgcn_global_load_async_to_lds_b128) && \
    __has_builtin(__builtin_amdgcn_s_wait_asynccnt)
#define HAVE_ASYNC 1
#else
#define HAVE_ASYNC 0
#endif

typedef int v4i __attribute__((vector_size(16)));   // 128-bit payload type
typedef __attribute__((address_space(1))) v4i gv4i; // global (AS1)
typedef __attribute__((address_space(3))) v4i lv4i; // LDS (AS3)

static __device__ __forceinline__ void async_cp16(const void* g, void* l) {
#if HAVE_ASYNC
    __builtin_amdgcn_global_load_async_to_lds_b128(
        (gv4i*)(unsigned long long)(uintptr_t)g,    // global src (AS1, v4i*)
        (lv4i*)(unsigned)(uintptr_t)l,              // LDS dst: low 32 bits = LDS offset
        0, 0);
#else
    *(float4*)l = *(const float4*)g;
#endif
}
#if HAVE_ASYNC
#define ASYNC_WAIT(n) __builtin_amdgcn_s_wait_asynccnt(n)
#else
#define ASYNC_WAIT(n)
#endif

// ---------------------------------------------------------------------------
// Kernel 1: fused 1x1-conv projections.
//   theta -> [B,N,F] f16, phi -> transposed [B,N,F] f16, g -> transposed [B,F,N] f16
// out[n,f] = sum_c x[b,c,n] * W[f,c] + bias[f];  WMMA M=n, N=f, K=c.
// ---------------------------------------------------------------------------
__device__ __forceinline__ void proj_one(
    const float* __restrict__ x, const float* __restrict__ W,
    const float* __restrict__ bias, _Float16* __restrict__ op,
    bool storeFN,                   // true: op[b][f][n]  (g), false: op[b][n][f]
    int b, int n0, int t, int w, int lm, int lh,
    _Float16 (*xs)[48], _Float16 (*wls)[48])
{
    v8f acc[8];
    #pragma unroll
    for (int i = 0; i < 8; ++i)
        #pragma unroll
        for (int j = 0; j < 8; ++j) acc[i][j] = 0.0f;

    for (int kc = 0; kc < 8; ++kc) {
        // stage x chunk [32 c][128 n] transposed -> xs[n][c] (f32 -> f16)
        #pragma unroll
        for (int i = 0; i < 16; ++i) {
            int idx = t + i * 256;
            int cl  = idx >> 7;
            int nl  = idx & 127;
            xs[nl][cl] = (_Float16)x[((size_t)b * CINX + kc * 32 + cl) * NNX + n0 + nl];
        }
        // stage W chunk [128 f][32 c]
        #pragma unroll
        for (int i = 0; i < 16; ++i) {
            int idx = t + i * 256;
            int f   = idx >> 5;
            int cl  = idx & 31;
            wls[f][cl] = (_Float16)W[(size_t)f * CINX + kc * 32 + cl];
        }
        __syncthreads();
        v16h afrag = *(const v16h*)&xs[w * 16 + lm][lh * 16];
        #pragma unroll
        for (int ft = 0; ft < 8; ++ft) {
            v16h bfrag = *(const v16h*)&wls[ft * 16 + lm][lh * 16];
            acc[ft] = __builtin_amdgcn_wmma_f32_16x16x32_f16(
                false, afrag, false, bfrag, (short)0, acc[ft], false, false);
        }
        __syncthreads();
    }
    // bias + store (C layout: VGPR r -> row n = w*16 + r + 8*half, col f = ft*16 + lm)
    #pragma unroll
    for (int ft = 0; ft < 8; ++ft) {
        float bv = bias[ft * 16 + lm];
        #pragma unroll
        for (int r = 0; r < 8; ++r) {
            int n = n0 + w * 16 + r + 8 * lh;
            int f = ft * 16 + lm;
            _Float16 v = (_Float16)(acc[ft][r] + bv);
            if (storeFN) op[((size_t)b * FFX + f) * NNX + n] = v;
            else         op[((size_t)b * NNX + n) * FFX + f] = v;
        }
    }
}

__global__ __launch_bounds__(256) void proj_kernel(
    const float* __restrict__ x,
    const float* __restrict__ Wg, const float* __restrict__ bg,
    const float* __restrict__ Wt, const float* __restrict__ bt,
    const float* __restrict__ Wp, const float* __restrict__ bp,
    _Float16* __restrict__ gT_h, _Float16* __restrict__ th_h,
    _Float16* __restrict__ phiT_h)
{
    __shared__ _Float16 xs[128][48];    // stride 96B (32B aligned)
    __shared__ _Float16 wls[128][48];

    const int bid = blockIdx.x;
    const int b = bid >> 5, nt = bid & 31, n0 = nt * 128;
    const int t = threadIdx.x, w = t >> 5, lane = t & 31;
    const int lm = lane & 15, lh = lane >> 4;

    proj_one(x, Wg, bg, gT_h,   true,  b, n0, t, w, lm, lh, xs, wls);
    __syncthreads();
    proj_one(x, Wt, bt, th_h,   false, b, n0, t, w, lm, lh, xs, wls);
    __syncthreads();
    proj_one(x, Wp, bp, phiT_h, false, b, n0, t, w, lm, lh, xs, wls);
}

// ---------------------------------------------------------------------------
// Kernel 2: flash attention with double-buffered async global->LDS staging.
// S = theta @ phi^T (K=F=128), online softmax, Y = P @ g.
// One block = 128 query rows; wave w owns 16 rows; 32 key columns per step.
// ---------------------------------------------------------------------------
__global__ __launch_bounds__(256) void attn_kernel(
    const _Float16* __restrict__ th_h, const _Float16* __restrict__ phiT_h,
    const _Float16* __restrict__ gT_h, _Float16* __restrict__ y_h)
{
    __shared__ _Float16 ps[2][32][144];    // phiT tile [m][f], row 288B
    __shared__ _Float16 gst[2][128][48];   // gT tile [f][m], row 96B
    __shared__ _Float16 plds[8][16][48];   // per-wave P transpose scratch

    const int bid = blockIdx.x;
    const int b = bid >> 5, nt = bid & 31, n0 = nt * 128;
    const int t = threadIdx.x, w = t >> 5, lane = t & 31;
    const int lm = lane & 15, lh = lane >> 4;

    // Preload theta A-fragments for this wave's 16 rows (K=128 -> 4 chunks)
    const int nrow = n0 + w * 16 + lm;
    v16h afr[4];
    #pragma unroll
    for (int kc = 0; kc < 4; ++kc)
        afr[kc] = *(const v16h*)&th_h[((size_t)b * NNX + nrow) * FFX + kc * 32 + lh * 16];

    v8f acc[8];
    #pragma unroll
    for (int i = 0; i < 8; ++i)
        #pragma unroll
        for (int j = 0; j < 8; ++j) acc[i][j] = 0.0f;
    float mrow[8], lrow[8];
    #pragma unroll
    for (int r = 0; r < 8; ++r) { mrow[r] = -3.0e38f; lrow[r] = 0.0f; }

    // stage one 32-column block into buffer nb (4 x b128 async per thread)
    auto stage = [&](int nb, int m0) {
        #pragma unroll
        for (int i = 0; i < 2; ++i) {       // phiT: 32 rows x 128 halves
            int e = t + i * 256;            // 0..511
            int m = e >> 4, c8 = e & 15;
            async_cp16(&phiT_h[((size_t)b * NNX + m0 + m) * FFX + c8 * 8],
                       &ps[nb][m][c8 * 8]);
        }
        #pragma unroll
        for (int i = 0; i < 2; ++i) {       // gT: 128 rows x 32 halves
            int e = t + i * 256;            // 0..511
            int f = e >> 2, c4 = e & 3;
            async_cp16(&gT_h[((size_t)b * FFX + f) * NNX + m0 + c4 * 8],
                       &gst[nb][f][c4 * 8]);
        }
    };

    const int NBLK = NNX / 32;              // 128
    stage(0, 0);                            // prologue prefetch

    for (int cb = 0; cb < NBLK; ++cb) {
        const int cur = cb & 1;
        __syncthreads();                    // all readers of buf[cur^1] (iter cb-1) done
        if (cb + 1 < NBLK) {
            stage((cb + 1) & 1, (cb + 1) * 32);
            ASYNC_WAIT(4);                  // in-order: all but newest 4 (= cur buffer) done
        } else {
            ASYNC_WAIT(0);
        }
        __syncthreads();                    // cur buffer visible to all waves

        // S tiles for 32 columns: s0 = cols 0..15, s1 = cols 16..31
        v8f s0, s1;
        #pragma unroll
        for (int j = 0; j < 8; ++j) { s0[j] = 0.0f; s1[j] = 0.0f; }
        #pragma unroll
        for (int kc = 0; kc < 4; ++kc) {
            v16h b0 = *(const v16h*)&ps[cur][lm][kc * 32 + lh * 16];
            v16h b1 = *(const v16h*)&ps[cur][16 + lm][kc * 32 + lh * 16];
            s0 = __builtin_amdgcn_wmma_f32_16x16x32_f16(false, afr[kc], false, b0,
                                                        (short)0, s0, false, false);
            s1 = __builtin_amdgcn_wmma_f32_16x16x32_f16(false, afr[kc], false, b1,
                                                        (short)0, s1, false, false);
        }
        // online softmax (row = (r, half); 16 cols across lanes of each half)
        #pragma unroll
        for (int r = 0; r < 8; ++r) {
            float mx = fmaxf(s0[r], s1[r]);
            #pragma unroll
            for (int m = 8; m >= 1; m >>= 1) mx = fmaxf(mx, __shfl_xor(mx, m, 32));
            float mnew  = fmaxf(mrow[r], mx);
            float alpha = __expf(mrow[r] - mnew);
            float p0 = __expf(s0[r] - mnew);
            float p1 = __expf(s1[r] - mnew);
            float rs = p0 + p1;
            #pragma unroll
            for (int m = 8; m >= 1; m >>= 1) rs += __shfl_xor(rs, m, 32);
            lrow[r] = lrow[r] * alpha + rs;
            mrow[r] = mnew;
            s0[r] = p0; s1[r] = p1;
            #pragma unroll
            for (int ft = 0; ft < 8; ++ft) acc[ft][r] *= alpha;
        }
        // transpose P (C layout -> A layout) through per-wave LDS scratch
        #pragma unroll
        for (int r = 0; r < 8; ++r) {
            int row = r + 8 * lh;
            plds[w][row][lm]      = (_Float16)s0[r];
            plds[w][row][16 + lm] = (_Float16)s1[r];
        }
        v16h pf = *(const v16h*)&plds[w][lm][lh * 16];   // same-wave RAW: dscnt wait
        #pragma unroll
        for (int ft = 0; ft < 8; ++ft) {
            v16h bf = *(const v16h*)&gst[cur][ft * 16 + lm][lh * 16];
            acc[ft] = __builtin_amdgcn_wmma_f32_16x16x32_f16(
                false, pf, false, bf, (short)0, acc[ft], false, false);
        }
    }

    // finalize: divide by row sums, store y f16 [B,N,F]
    #pragma unroll
    for (int ft = 0; ft < 8; ++ft) {
        #pragma unroll
        for (int r = 0; r < 8; ++r) {
            int n = n0 + w * 16 + r + 8 * lh;
            int f = ft * 16 + lm;
            y_h[((size_t)b * NNX + n) * FFX + f] = (_Float16)(acc[ft][r] / lrow[r]);
        }
    }
}

// ---------------------------------------------------------------------------
// Kernel 3: output projection z[b,c,n] = Wz[c,:] . y[n,:] + bz[c].
// Writes pre-norm z to d_out + deterministic per-block (sum,sumsq) partials.
// ---------------------------------------------------------------------------
__global__ __launch_bounds__(256) void zproj_kernel(
    const _Float16* __restrict__ y_h, const float* __restrict__ Wz,
    const float* __restrict__ bz, float* __restrict__ zout,
    float* __restrict__ part)
{
    __shared__ _Float16 wzs[128][80];   // [c][f half], row 160B
    __shared__ _Float16 ys[128][80];    // [n][f half]
    __shared__ float redsum[8], redsq[8];

    const int bid = blockIdx.x;
    const int b  = bid >> 6;
    const int wi = bid & 63;
    const int c0 = (wi >> 5) * 128;
    const int n0 = (wi & 31) * 128;
    const int t = threadIdx.x, w = t >> 5, lane = t & 31;
    const int lm = lane & 15, lh = lane >> 4;

    v8f acc[8];
    #pragma unroll
    for (int i = 0; i < 8; ++i)
        #pragma unroll
        for (int j = 0; j < 8; ++j) acc[i][j] = 0.0f;

    for (int fh = 0; fh < 2; ++fh) {    // K = 128 in two 64 halves
        // async stage y tile (f16, no conversion needed)
        #pragma unroll
        for (int i = 0; i < 4; ++i) {
            int e = t + i * 256;        // 0..1023
            int r = e >> 3, f8 = e & 7; // 128 rows x 8 chunks of 8 halves
            async_cp16(&y_h[((size_t)b * NNX + n0 + r) * FFX + fh * 64 + f8 * 8],
                       &ys[r][f8 * 8]);
        }
        // manual stage Wz tile (f32 -> f16 conversion)
        #pragma unroll
        for (int i = 0; i < 32; ++i) {
            int e = t + i * 256;
            int r = e >> 6, f = e & 63;
            wzs[r][f] = (_Float16)Wz[(size_t)(c0 + r) * FFX + fh * 64 + f];
        }
        ASYNC_WAIT(0);
        __syncthreads();
        #pragma unroll
        for (int kc2 = 0; kc2 < 2; ++kc2) {
            v16h af = *(const v16h*)&wzs[w * 16 + lm][kc2 * 32 + lh * 16];
            #pragma unroll
            for (int ntile = 0; ntile < 8; ++ntile) {
                v16h bf = *(const v16h*)&ys[ntile * 16 + lm][kc2 * 32 + lh * 16];
                acc[ntile] = __builtin_amdgcn_wmma_f32_16x16x32_f16(
                    false, af, false, bf, (short)0, acc[ntile], false, false);
            }
        }
        __syncthreads();
    }

    float ls = 0.0f, lq = 0.0f;
    const int crow = c0 + w * 16;
    #pragma unroll
    for (int ntile = 0; ntile < 8; ++ntile) {
        #pragma unroll
        for (int r = 0; r < 8; ++r) {
            int c = crow + r + 8 * lh;
            int n = n0 + ntile * 16 + lm;
            float v = acc[ntile][r] + bz[c];
            zout[((size_t)b * CINX + c) * NNX + n] = v;
            ls += v; lq += v * v;
        }
    }
    #pragma unroll
    for (int m = 16; m >= 1; m >>= 1) {
        ls += __shfl_xor(ls, m, 32);
        lq += __shfl_xor(lq, m, 32);
    }
    if (lane == 0) { redsum[w] = ls; redsq[w] = lq; }
    __syncthreads();
    if (t == 0) {
        float s = 0.0f, q = 0.0f;
        #pragma unroll
        for (int i = 0; i < 8; ++i) { s += redsum[i]; q += redsq[i]; }
        part[(b * 64 + wi) * 2 + 0] = s;
        part[(b * 64 + wi) * 2 + 1] = q;
    }
}

// ---------------------------------------------------------------------------
// Kernel 4: per-batch mean / rstd from 64 block partials.
// ---------------------------------------------------------------------------
__global__ void stats_kernel(const float* __restrict__ part, float* __restrict__ stats)
{
    int b = blockIdx.x;
    if (threadIdx.x == 0) {
        float s = 0.0f, q = 0.0f;
        for (int i = 0; i < 64; ++i) {
            s += part[(b * 64 + i) * 2 + 0];
            q += part[(b * 64 + i) * 2 + 1];
        }
        float inv = 1.0f / ((float)CINX * (float)NNX);
        float mu  = s * inv;
        float var = q * inv - mu * mu;
        stats[b * 2 + 0] = mu;
        stats[b * 2 + 1] = rsqrtf(var + 1e-5f);
    }
}

// ---------------------------------------------------------------------------
// Kernel 5: in-place LayerNorm affine + residual (float4 vectorized).
// ---------------------------------------------------------------------------
__global__ __launch_bounds__(256) void norm_kernel(
    const float* __restrict__ x, const float* __restrict__ lnw,
    const float* __restrict__ lnb, const float* __restrict__ stats,
    float* __restrict__ out)
{
    const int i = blockIdx.x * blockDim.x + threadIdx.x;
    const int cn_total = CINX * NNX;
    int elem = i * 4;
    int b  = elem / cn_total;
    int cn = elem - b * cn_total;
    float mu   = stats[b * 2 + 0];
    float rstd = stats[b * 2 + 1];
    float4 z  = ((const float4*)out)[i];
    float4 xv = ((const float4*)x)[i];
    float4 wv = ((const float4*)lnw)[cn >> 2];
    float4 bv = ((const float4*)lnb)[cn >> 2];
    float4 o;
    o.x = (z.x - mu) * rstd * wv.x + bv.x + xv.x;
    o.y = (z.y - mu) * rstd * wv.y + bv.y + xv.y;
    o.z = (z.z - mu) * rstd * wv.z + bv.z + xv.z;
    o.w = (z.w - mu) * rstd * wv.w + bv.w + xv.w;
    ((float4*)out)[i] = o;
}

// ---------------------------------------------------------------------------
extern "C" void kernel_launch(void* const* d_in, const int* in_sizes, int n_in,
                              void* d_out, int out_size, void* d_ws, size_t ws_size,
                              hipStream_t stream)
{
    const float* x    = (const float*)d_in[0];
    const float* Wg   = (const float*)d_in[1];
    const float* bg   = (const float*)d_in[2];
    const float* Wt   = (const float*)d_in[3];
    const float* bt   = (const float*)d_in[4];
    const float* Wp   = (const float*)d_in[5];
    const float* bp   = (const float*)d_in[6];
    const float* Wz   = (const float*)d_in[7];
    const float* bz   = (const float*)d_in[8];
    const float* ln_w = (const float*)d_in[9];
    const float* ln_b = (const float*)d_in[10];

    const size_t S = (size_t)BB * NNX * FFX;       // 2,097,152 halves each
    _Float16* base   = (_Float16*)d_ws;
    _Float16* gT_h   = base;                        // [B,F,N]
    _Float16* th_h   = base + S;                    // [B,N,F]
    _Float16* phiT_h = base + 2 * S;                // [B,N,F]
    _Float16* y_h    = base + 3 * S;                // [B,N,F]
    float* part  = (float*)(base + 4 * S);          // B*64*2 floats
    float* stats = part + BB * 64 * 2;              // B*2 floats
    float* zout  = (float*)d_out;

    proj_kernel <<<BB * 32, 256, 0, stream>>>(x, Wg, bg, Wt, bt, Wp, bp,
                                              gT_h, th_h, phiT_h);
    attn_kernel <<<BB * 32, 256, 0, stream>>>(th_h, phiT_h, gT_h, y_h);
    zproj_kernel<<<BB * 64, 256, 0, stream>>>(y_h, Wz, bz, zout, part);
    stats_kernel<<<BB, 64, 0, stream>>>(part, stats);
    norm_kernel <<<(BB * CINX * NNX) / (4 * 256), 256, 0, stream>>>(x, ln_w, ln_b,
                                                                    stats, zout);
}
